// ResidualGCNEncoder_67534065762493
// MI455X (gfx1250) — compile-verified, matrix operand order checked by reading, not run on hardware
//
#include <hip/hip_runtime.h>
#include <hip/hip_bf16.h>

#define N_NODES 50000
#define N_EDGES 800000
#define D_IN 128
#define D_H  256
#define LN_EPS 1e-5f

typedef __attribute__((ext_vector_type(16))) __bf16 bf16x16;
typedef __attribute__((ext_vector_type(8)))  __bf16 bf16x8;
typedef __attribute__((ext_vector_type(8)))  float  f32x8;

__device__ __forceinline__ __bf16 to_bf16(float x)  { return (__bf16)x; }
__device__ __forceinline__ __bf16 to_bf16(__bf16 x) { return x; }

// ---------------------------------------------------------------- degree / dinv
__global__ void deg_kernel(const int* __restrict__ dst, float* __restrict__ deg, int ne) {
    int e = blockIdx.x * blockDim.x + threadIdx.x;
    if (e < ne) atomicAdd(&deg[dst[e]], 1.0f);
}

__global__ void dinv_kernel(float* __restrict__ deg, int n) {
    int i = blockIdx.x * blockDim.x + threadIdx.x;
    if (i < n) deg[i] = rsqrtf(deg[i] + 1.0f);   // +1 for the self-loop; always > 0
}

// ---------------------------------------------------------------- WMMA GEMM
// H[nrows x NOUT] = A[nrows x K] @ W[K x NOUT], bf16 inputs, fp32 accumulate.
// 8 waves / block; each wave computes one 16x16 tile of H.
// W's 16-column tile is staged in LDS pre-swizzled into WMMA B-fragment order
// ([K/32 chunks][32 lanes][16 bf16]) so the inner-loop B load per lane is one
// contiguous 32-byte read (2x ds_load_b128).
template<int K, int NOUT, typename AT>
__global__ void __launch_bounds__(256)
gemm_wmma_kernel(const AT* __restrict__ A, const float* __restrict__ W,
                 float* __restrict__ H, int nrows) {
    __shared__ alignas(32) __bf16 lds_b[K * 16];

    const int n0 = blockIdx.y * 16;

    // Stage W tile: one unit = 8 consecutive fragment elements of one lane
    // (one 16-byte ds_store_b128). Units: (K/32 chunks) * 32 lanes * 2 runs.
    for (int u = threadIdx.x; u < K * 2; u += 256) {
        const int chunk = u >> 6;           // 64 units per K-chunk
        const int rem   = u & 63;
        const int l     = rem >> 1;         // consuming lane
        const int run   = rem & 1;          // elems 0..7 or 8..15
        const int lhalf = l >> 4;
        const int mr    = l & 15;
        // 16-bit A/B fragment K mapping (ISA 7.12.2):
        //   lane<16: elems 0..7 -> K+0..7, elems 8..15 -> K+16..23; lane>=16: +8
        const int kbase = chunk * 32 + run * 16 + lhalf * 8;
        bf16x8 v{};
#pragma unroll
        for (int e = 0; e < 8; ++e)
            v[e] = to_bf16(W[(size_t)(kbase + e) * NOUT + n0 + mr]);
        *(bf16x8*)&lds_b[(chunk * 32 + l) * 16 + run * 8] = v;
    }
    __syncthreads();

    const int lane = threadIdx.x & 31;
    const int wave = threadIdx.x >> 5;
    const int half = lane >> 4;
    const int mr   = lane & 15;
    const int m0   = (blockIdx.x * 8 + wave) * 16;
    if (m0 >= nrows) return;

    f32x8 acc = {};
    const AT* __restrict__ arow = A + (size_t)(m0 + mr) * K;
    const bf16x16* __restrict__ bfrag = (const bf16x16*)lds_b;

#pragma unroll
    for (int kk = 0; kk < K; kk += 32) {
        bf16x16 a;
#pragma unroll
        for (int e = 0; e < 8; ++e) {
            a[e]     = to_bf16(arow[kk + half * 8 + e]);        // contiguous: b128 loads
            a[8 + e] = to_bf16(arow[kk + 16 + half * 8 + e]);
        }
        const bf16x16 b = bfrag[(kk >> 5) * 32 + lane];          // 2x ds_load_b128
        acc = __builtin_amdgcn_wmma_f32_16x16x32_bf16(
                false, a, false, b, (short)0, acc, false, false);
    }

    // C/D 16x16 f32 layout: VGPR v holds row M = v + 8*half, col = mr
#pragma unroll
    for (int v = 0; v < 8; ++v) {
        H[(size_t)(m0 + v + 8 * half) * NOUT + n0 + mr] = acc[v];
    }
}

// ---------------------------------------------------------------- edge scatter
// Dominant kernel: gather h[src] row (L2-resident), scale by norm, scatter-add
// to agg[dst] with fp32 atomics. float4 per thread (global_load_b128), several
// edges per 256-thread block to keep WGPs saturated with 8 waves.
template<int D>
__global__ void __launch_bounds__(256)
aggregate_kernel(const float* __restrict__ h, const int* __restrict__ src,
                 const int* __restrict__ dst, const float* __restrict__ dinv,
                 float* __restrict__ agg, int ne) {
    constexpr int TPE = D / 4;                     // threads per edge
    constexpr int EPB = 256 / TPE;                 // edges per block
    const int e = blockIdx.x * EPB + threadIdx.x / TPE;
    if (e >= ne) return;
    const int s = src[e];
    const int d = dst[e];
    const int c = (threadIdx.x % TPE) * 4;
    const float4* __restrict__ hs = (const float4*)(h + (size_t)s * D + c);
    __builtin_prefetch(hs, 0, 3);                  // global_prefetch_b8
    const float nrm = dinv[s] * dinv[d];
    const float4 v = *hs;
    float* __restrict__ ad = agg + (size_t)d * D + c;
    atomicAdd(ad + 0, v.x * nrm);
    atomicAdd(ad + 1, v.y * nrm);
    atomicAdd(ad + 2, v.z * nrm);
    atomicAdd(ad + 3, v.w * nrm);
}

// ---------------------------------------------------------------- reductions
template<int BS>
__device__ __forceinline__ float block_sum(float x, float* sbuf) {
#pragma unroll
    for (int o = 16; o > 0; o >>= 1) x += __shfl_xor(x, o, 32);   // wave32 reduce
    const int wid = threadIdx.x >> 5;
    if ((threadIdx.x & 31) == 0) sbuf[wid] = x;
    __syncthreads();
    float tot = 0.0f;
#pragma unroll
    for (int w = 0; w < BS / 32; ++w) tot += sbuf[w];
    __syncthreads();
    return tot;
}

// ---------------------------------------------------------------- epilogue 1
// val = agg + h*dinv^2 (self loop) + b1 -> LayerNorm -> erf-GELU -> bf16 hidden
__global__ void __launch_bounds__(D_H)
post1_kernel(const float* __restrict__ agg, const float* __restrict__ h,
             const float* __restrict__ dinv, const float* __restrict__ b1,
             const float* __restrict__ g1,  const float* __restrict__ be1,
             __bf16* __restrict__ hidden) {
    __shared__ float sbuf[D_H / 32];
    const int i = blockIdx.x;
    const int c = threadIdx.x;
    const float di = dinv[i];
    float v = agg[(size_t)i * D_H + c] + h[(size_t)i * D_H + c] * di * di + b1[c];

    const float mu  = block_sum<D_H>(v, sbuf) * (1.0f / D_H);
    const float d   = v - mu;
    const float var = block_sum<D_H>(d * d, sbuf) * (1.0f / D_H);
    float y = d * rsqrtf(var + LN_EPS) * g1[c] + be1[c];
    y = 0.5f * y * (1.0f + erff(y * 0.70710678118654752f));       // exact GELU
    hidden[(size_t)i * D_H + c] = (__bf16)y;
}

// ---------------------------------------------------------------- epilogue 2
// val = agg + h2*dinv^2 + b2 -> LN -> res = x + val -> L2 normalize -> out
__global__ void __launch_bounds__(D_IN)
final_kernel(const float* __restrict__ agg, const float* __restrict__ h2,
             const float* __restrict__ dinv, const float* __restrict__ b2,
             const float* __restrict__ g2,  const float* __restrict__ be2,
             const float* __restrict__ x,   float* __restrict__ out) {
    __shared__ float sbuf[D_IN / 32];
    const int i = blockIdx.x;
    const int c = threadIdx.x;
    const float di = dinv[i];
    float v = agg[(size_t)i * D_IN + c] + h2[(size_t)i * D_IN + c] * di * di + b2[c];

    const float mu  = block_sum<D_IN>(v, sbuf) * (1.0f / D_IN);
    const float d   = v - mu;
    const float var = block_sum<D_IN>(d * d, sbuf) * (1.0f / D_IN);
    const float y   = d * rsqrtf(var + LN_EPS) * g2[c] + be2[c];

    const float r  = x[(size_t)i * D_IN + c] + y;
    const float ss = block_sum<D_IN>(r * r, sbuf);
    const float nrm = sqrtf(ss);
    out[(size_t)i * D_IN + c] = r / fmaxf(nrm, 1e-12f);
}

// ---------------------------------------------------------------- launcher
extern "C" void kernel_launch(void* const* d_in, const int* in_sizes, int n_in,
                              void* d_out, int out_size, void* d_ws, size_t ws_size,
                              hipStream_t stream) {
    const float* x   = (const float*)d_in[0];
    const int*   ei  = (const int*)  d_in[1];   // [2, E] int32
    const float* W1  = (const float*)d_in[2];
    const float* b1  = (const float*)d_in[3];
    const float* W2  = (const float*)d_in[4];
    const float* b2  = (const float*)d_in[5];
    const float* g1  = (const float*)d_in[6];
    const float* be1 = (const float*)d_in[7];
    const float* g2  = (const float*)d_in[8];
    const float* be2 = (const float*)d_in[9];
    const int* src = ei;
    const int* dst = ei + N_EDGES;

    char* ws = (char*)d_ws;
    size_t off = 0;
    float*  dinv = (float*)(ws + off);  off += (size_t)N_NODES * 4;
    float*  h    = (float*)(ws + off);  off += (size_t)N_NODES * D_H * 4;   // h1 / h2
    float*  agg  = (float*)(ws + off);  off += (size_t)N_NODES * D_H * 4;   // agg1 / agg2
    __bf16* hid  = (__bf16*)(ws + off); off += (size_t)N_NODES * D_H * 2;   // bf16 hidden

    // degree -> dinv
    (void)hipMemsetAsync(dinv, 0, (size_t)N_NODES * 4, stream);
    deg_kernel<<<(N_EDGES + 255) / 256, 256, 0, stream>>>(dst, dinv, N_EDGES);
    dinv_kernel<<<(N_NODES + 255) / 256, 256, 0, stream>>>(dinv, N_NODES);

    // ---- layer 1: h = x @ W1 ; scatter ; LN + GELU -> bf16 hidden
    dim3 grid1((N_NODES / 16 + 7) / 8, D_H / 16);
    gemm_wmma_kernel<D_IN, D_H, float><<<grid1, 256, 0, stream>>>(x, W1, h, N_NODES);
    (void)hipMemsetAsync(agg, 0, (size_t)N_NODES * D_H * 4, stream);
    {   // 4 edges / block (64 threads per edge, float4 each)
        const int epb = 256 / (D_H / 4);
        aggregate_kernel<D_H><<<(N_EDGES + epb - 1) / epb, 256, 0, stream>>>(
            h, src, dst, dinv, agg, N_EDGES);
    }
    post1_kernel<<<N_NODES, D_H, 0, stream>>>(agg, h, dinv, b1, g1, be1, hid);

    // ---- layer 2: h2 = hidden @ W2 ; scatter ; LN + residual + L2 normalize
    dim3 grid2((N_NODES / 16 + 7) / 8, D_IN / 16);
    gemm_wmma_kernel<D_H, D_IN, __bf16><<<grid2, 256, 0, stream>>>(hid, W2, h, N_NODES);
    (void)hipMemsetAsync(agg, 0, (size_t)N_NODES * D_IN * 4, stream);
    {   // 8 edges / block (32 threads per edge, float4 each)
        const int epb = 256 / (D_IN / 4);
        aggregate_kernel<D_IN><<<(N_EDGES + epb - 1) / epb, 256, 0, stream>>>(
            h, src, dst, dinv, agg, N_EDGES);
    }
    final_kernel<<<N_NODES, D_IN, 0, stream>>>(agg, h, dinv, b2, g2, be2, x, (float*)d_out);
}
